// SmoothLinearQuantizer_82489141887066
// MI455X (gfx1250) — 8-bit matrix operand order repaired and checked by reading, compiled, not run on hardware
//
#include <hip/hip_runtime.h>
#include <stdint.h>

typedef __attribute__((ext_vector_type(8))) int v8i;

#define QMAX 127.0f

__global__ __launch_bounds__(256)
void quant_rows(const float* __restrict__ in, int8_t* __restrict__ q,
                float* __restrict__ scales, int K) {
    const int row = blockIdx.x;
    const int t   = threadIdx.x;
    const float4* r4 = (const float4*)(in + (size_t)row * K);

    float4 vals[4];
    float m = 0.0f;
#pragma unroll
    for (int i = 0; i < 4; ++i) {
        vals[i] = r4[t + i * 256];
        m = fmaxf(m, fmaxf(fmaxf(fabsf(vals[i].x), fabsf(vals[i].y)),
                           fmaxf(fabsf(vals[i].z), fabsf(vals[i].w))));
    }

    __shared__ float red[256];
    red[t] = m;
    __syncthreads();
#pragma unroll
    for (int s = 128; s > 0; s >>= 1) {
        if (t < s) red[t] = fmaxf(red[t], red[t + s]);
        __syncthreads();
    }
    const float scale = fmaxf(red[0], 1e-5f) * (1.0f / QMAX);
    const float inv   = 1.0f / scale;
    if (t == 0) scales[row] = scale;

    int8_t* qr = q + (size_t)row * K;
#pragma unroll
    for (int i = 0; i < 4; ++i) {
        float4 v = vals[i];
        char4 c;
        c.x = (char)__float2int_rn(v.x * inv);
        c.y = (char)__float2int_rn(v.y * inv);
        c.z = (char)__float2int_rn(v.z * inv);
        c.w = (char)__float2int_rn(v.w * inv);
        *(char4*)(qr + (size_t)t * 4 + (size_t)i * 1024) = c;
    }
}

__global__ __launch_bounds__(256)
void gemm_i8_wmma(const int8_t* __restrict__ qx, const int8_t* __restrict__ qw,
                  const float* __restrict__ sx, const float* __restrict__ sw,
                  const float* __restrict__ bias, float* __restrict__ y,
                  int N, int K) {
    const int lane = threadIdx.x & 31;
    const int wave = threadIdx.x >> 5;
    const int half = lane >> 4;
    const int l15  = lane & 15;
    const int m0   = blockIdx.x * 256 + (wave & 3) * 64;
    const int n0   = blockIdx.y * 64  + (wave >> 2) * 32;

    v8i acc[4][2] = {};

    uint32_t aOff[4];
#pragma unroll
    for (int ms = 0; ms < 4; ++ms)
        aOff[ms] = (uint32_t)(m0 + ms * 16 + l15) * (uint32_t)K + half * 16;
    uint32_t bOff[2];
#pragma unroll
    for (int ns = 0; ns < 2; ++ns)
        bOff[ns] = (uint32_t)(n0 + ns * 16 + l15) * (uint32_t)K + half * 16;

    for (int k0 = 0; k0 < K; k0 += 256) {
        const int8_t* qxk = qx + k0;
        const int8_t* qwk = qw + k0;
#pragma unroll
        for (int kk = 0; kk < 256; kk += 64) {
            __builtin_prefetch(qxk + aOff[0] + kk + 256, 0, 3);
            __builtin_prefetch(qwk + bOff[0] + kk + 256, 0, 3);

            v8i a[4], b[2];
#pragma unroll
            for (int ms = 0; ms < 4; ++ms) {
                const int2* p = (const int2*)(qxk + aOff[ms] + kk);
                int2 t0 = p[0], t1 = p[1], t2 = p[4], t3 = p[5];
                v8i av = {t0.x, t0.y, t1.x, t1.y, t2.x, t2.y, t3.x, t3.y};
                a[ms] = av;
            }
#pragma unroll
            for (int ns = 0; ns < 2; ++ns) {
                const int4* p = (const int4*)(qwk + bOff[ns] + kk);
                int4 u0 = p[0], u1 = p[2];
                v8i bv = {u0.x, u0.y, u0.z, u0.w, u1.x, u1.y, u1.z, u1.w};
                b[ns] = bv;
            }
#pragma unroll
            for (int ms = 0; ms < 4; ++ms)
#pragma unroll
                for (int ns = 0; ns < 2; ++ns)
                    acc[ms][ns] = __builtin_amdgcn_wmma_i32_16x16x64_iu8(
                         true, a[ms],  true, b[ns],
                        acc[ms][ns],  false,  false);
        }
    }

#pragma unroll
    for (int ns = 0; ns < 2; ++ns) {
        const int n    = n0 + ns * 16 + l15;
        const float wn = sw[n];
        const float bn = bias[n];
#pragma unroll
        for (int ms = 0; ms < 4; ++ms) {
            const int mb = m0 + ms * 16 + half * 8;
#pragma unroll
            for (int v = 0; v < 8; ++v) {
                const int m = mb + v;
                y[(size_t)m * N + n] = sx[m] * wn * (float)acc[ms][ns][v] + bn;
            }
        }
    }
}

__global__ __launch_bounds__(256)
void quant_out_inplace(float* __restrict__ y, int N) {
    const int row = blockIdx.x;
    const int t   = threadIdx.x;
    float4* r4 = (float4*)(y + (size_t)row * N);

    float4 vals[4];
    float m = 0.0f;
#pragma unroll
    for (int i = 0; i < 4; ++i) {
        vals[i] = r4[t + i * 256];
        m = fmaxf(m, fmaxf(fmaxf(fabsf(vals[i].x), fabsf(vals[i].y)),
                           fmaxf(fabsf(vals[i].z), fabsf(vals[i].w))));
    }

    __shared__ float red[256];
    red[t] = m;
    __syncthreads();
#pragma unroll
    for (int s = 128; s > 0; s >>= 1) {
        if (t < s) red[t] = fmaxf(red[t], red[t + s]);
        __syncthreads();
    }
    const float scale = fmaxf(red[0], 1e-5f) * (1.0f / QMAX);
    const float inv   = 1.0f / scale;

#pragma unroll
    for (int i = 0; i < 4; ++i) {
        float4 v = vals[i];
        v.x = rintf(v.x * inv) * scale;
        v.y = rintf(v.y * inv) * scale;
        v.z = rintf(v.z * inv) * scale;
        v.w = rintf(v.w * inv) * scale;
        r4[t + i * 256] = v;
    }
}

extern "C" void kernel_launch(void* const* d_in, const int* in_sizes, int n_in,
                              void* d_out, int out_size, void* d_ws, size_t ws_size,
                              hipStream_t stream) {
    const int M = 4 * 2048;
    const int N = 4096;
    const int K = 4096;

    const float* input  = (const float*)d_in[0];
    const float* weight = (const float*)d_in[1];
    const float* bias   = (const float*)d_in[2];
    float*       out    = (float*)d_out;

    int8_t* qx = (int8_t*)d_ws;
    int8_t* qw = qx + (size_t)M * K;
    float*  sx = (float*)(qw + (size_t)N * K);
    float*  sw = sx + M;

    quant_rows<<<N, 256, 0, stream>>>(weight, qw, sw, K);
    quant_rows<<<M, 256, 0, stream>>>(input,  qx, sx, K);

    dim3 grid(M / 256, N / 64);
    gemm_i8_wmma<<<grid, 256, 0, stream>>>(qx, qw, sx, sw, bias, out, N, K);

    quant_out_inplace<<<M, 256, 0, stream>>>(out, N);
}
